// ConvEncoder_44985487458668
// MI455X (gfx1250) — compile-verified, matrix operand order checked by reading, not run on hardware
//
#include <hip/hip_runtime.h>

// ---------- types ----------
typedef __attribute__((ext_vector_type(16))) __bf16 v16bf;
typedef __attribute__((ext_vector_type(8)))  float  v8f;

#define DEV static __device__ __forceinline__

// float -> bf16 bits, round-to-nearest-even
DEV unsigned short f2bf(float f) {
  unsigned u = __float_as_uint(f);
  unsigned r = u + 0x7FFFu + ((u >> 16) & 1u);
  return (unsigned short)(r >> 16);
}

// Load one 16x32 bf16 fragment (A-layout): lanes 0-15 rows M, half-wave picks
// K-phase; per lane two contiguous 8-bf16 (16B) runs -> two global_load_b128.
DEV v16bf load_frag(const unsigned short* base, int ld, int row0, int k0, int lane) {
  int r  = row0 + (lane & 15);
  int kb = k0 + ((lane & 16) ? 8 : 0);
  const unsigned short* p = base + (size_t)r * ld + kb;
  union { uint4 q[2]; v16bf v; } f;
  f.q[0] = *reinterpret_cast<const uint4*>(p);
  f.q[1] = *reinterpret_cast<const uint4*>(p + 16);
  return f.v;
}

DEV v16bf load_frag_row(const unsigned short* rowptr, int k0, int lane) {
  int kb = k0 + ((lane & 16) ? 8 : 0);
  const unsigned short* p = rowptr + kb;
  union { uint4 q[2]; v16bf v; } f;
  f.q[0] = *reinterpret_cast<const uint4*>(p);
  f.q[1] = *reinterpret_cast<const uint4*>(p + 16);
  return f.v;
}

DEV v8f wmma_bf16(v16bf a, v16bf b, v8f c) {
  return __builtin_amdgcn_wmma_f32_16x16x32_bf16(false, a, false, b, (short)0, c, false, false);
}

// ---------- output offsets (flat float d_out) ----------
#define O_Z2M  0
#define O_Z2LV 10240
#define O_YL   20480
#define O_YI   53248
#define O_YH   57344
#define O_ZM   90112
#define O_ZLV  94208
#define O_ZS   98304

// ================= weight prep =================
__global__ void wprep_conv_k(const float* w, unsigned short* dst, int OC, int K, int Kpad) {
  int i = blockIdx.x * blockDim.x + threadIdx.x;
  if (i >= OC * Kpad) return;
  int k = i % Kpad, oc = i / Kpad;
  dst[i] = f2bf(k < K ? w[(size_t)oc * K + k] : 0.f);
}

// sw_w1 (S,NB,ND,NDS) -> Bt1 (S,NB,NDS,ND)
__global__ void w1prep_k(const float* w, unsigned short* dst) {
  long long i = blockIdx.x * (long long)blockDim.x + threadIdx.x;
  if (i >= (long long)4 * 8 * 256 * 1024) return;
  int d = (int)(i % 1024); long long t = i / 1024;
  int h = (int)(t % 256);  int sn = (int)(t / 256);
  dst[i] = f2bf(w[((size_t)sn * 1024 + d) * 256 + h]);
}

// sw_w2 (S,NB,NDS,ND) -> Bt2 (S,NB,ND,NDS)
__global__ void w2prep_k(const float* w, unsigned short* dst) {
  long long i = blockIdx.x * (long long)blockDim.x + threadIdx.x;
  if (i >= (long long)4 * 8 * 1024 * 256) return;
  int h = (int)(i % 256); long long t = i / 256;
  int d = (int)(t % 1024); int sn = (int)(t / 1024);
  dst[i] = f2bf(w[((size_t)sn * 256 + h) * 1024 + d]);
}

// ================= im2col (stride 2, pad 1, 4x4) =================
__global__ void im2col_k(const float* __restrict__ src, unsigned short* __restrict__ patch,
                         int C, int H, int W, int OH, int OW, int Kpad) {
  long long idx = blockIdx.x * (long long)blockDim.x + threadIdx.x;
  long long total = (long long)1024 * OH * OW * Kpad;
  if (idx >= total) return;
  int k = (int)(idx % Kpad);
  long long m = idx / Kpad;
  int ow = (int)(m % OW); long long t = m / OW;
  int oh = (int)(t % OH); int b = (int)(t / OH);
  int K = C * 16;
  float val = 0.f;
  if (k < K) {
    int kw = k & 3, kh = (k >> 2) & 3, c = k >> 4;
    int ih = oh * 2 - 1 + kh, iw = ow * 2 - 1 + kw;
    if (ih >= 0 && ih < H && iw >= 0 && iw < W)
      val = src[(((size_t)b * C + c) * H + ih) * W + iw];
  }
  patch[idx] = f2bf(val);
}

// ================= conv GEMM: (M x Kpad)bf16 @ (Kpad x OC)bf16 -> NCHW f32 =================
// block = 256 thr = 8 waves; wave owns a 16-row tile and ALL N-tiles (register-
// blocked: A fragment loaded once per K-step, NT back-to-back WMMAs).
template <int NT>
__global__ void conv_gemm_k(const unsigned short* __restrict__ A,
                            const unsigned short* __restrict__ Wt,   // (OC x Kpad) bf16
                            const float* __restrict__ bias,
                            float* __restrict__ dst, int Kpad, int OHW, int relu) {
  constexpr int OC = NT * 16;
  int lane = threadIdx.x & 31;
  int wv   = threadIdx.x >> 5;
  long long row0 = ((long long)blockIdx.x * 8 + wv) * 16;
  v8f acc[NT];
#pragma unroll
  for (int i = 0; i < NT; ++i) acc[i] = {};
  for (int k0 = 0; k0 < Kpad; k0 += 32) {
    __builtin_prefetch((const void*)(A + (size_t)row0 * Kpad + k0 + 256), 0, 3);
    v16bf a = load_frag(A, Kpad, (int)row0, k0, lane);
#pragma unroll
    for (int nt = 0; nt < NT; ++nt) {
      v16bf b = load_frag(Wt, Kpad, nt * 16, k0, lane);
      acc[nt] = wmma_bf16(a, b, acc[nt]);
    }
  }
  long long mb = row0 + ((lane & 16) ? 8 : 0);
#pragma unroll
  for (int nt = 0; nt < NT; ++nt) {
    int col = nt * 16 + (lane & 15);
    float bs = bias[col];
#pragma unroll
    for (int v = 0; v < 8; ++v) {
      long long m = mb + v;
      int ohw = (int)(m % OHW);
      long long bb = m / OHW;
      float x = acc[nt][v] + bs;
      if (relu) x = fmaxf(x, 0.f);
      dst[(bb * OC + col) * (long long)OHW + ohw] = x;
    }
  }
}

// ================= control + gumbel/softmax/argmax (per sample) =================
__global__ void ctrl_k(const float* __restrict__ outbuf,
                       const float* __restrict__ sw_w, const float* __restrict__ sw_b,
                       const float* __restrict__ gu, const float* __restrict__ gn,
                       float* __restrict__ dout, int layer,
                       int* __restrict__ nsel, float* __restrict__ zsel) {
  __shared__ float h[1024];
  __shared__ float ctrl[24];
  int b = blockIdx.x;
  for (int t = threadIdx.x; t < 1024; t += blockDim.x)
    h[t] = fmaxf(outbuf[(size_t)b * 1024 + t], 0.f);
  __syncthreads();
  int j = threadIdx.x;
  if (j < 24) {
    const float* wc = sw_w + (size_t)layer * 1024 * 24 + j;
    float s = sw_b[layer * 24 + j];
    for (int d = 0; d < 1024; ++d) s += h[d] * wc[(size_t)d * 24];
    ctrl[j] = s;
  }
  __syncthreads();
  if (threadIdx.x == 0) {
    float yl[8], zm[8], zlv[8], g[8], ys[8], z[8];
    float mx = -1e30f;
    for (int n = 0; n < 8; ++n) {
      yl[n] = ctrl[n]; zm[n] = ctrl[8 + n]; zlv[n] = ctrl[16 + n];
      float u = gu[((size_t)layer * 1024 + b) * 8 + n];
      float e = -log1pf(-u);
      g[n] = yl[n] - logf(e + 1e-20f);
      mx = fmaxf(mx, g[n]);
    }
    float se = 0.f;
    for (int n = 0; n < 8; ++n) { ys[n] = expf(g[n] - mx); se += ys[n]; }
    int am = 0; float best = -1.f;
    for (int n = 0; n < 8; ++n) { ys[n] /= se; if (ys[n] > best) { best = ys[n]; am = n; } }
    for (int n = 0; n < 8; ++n)
      z[n] = gn[((size_t)layer * 1024 + b) * 8 + n] * expf(zlv[n] * 0.5f) + zm[n];
    float* p = dout + O_YL + ((size_t)layer * 1024 + b) * 8;
    for (int n = 0; n < 8; ++n) p[n] = yl[n];
    dout[O_YI + (size_t)layer * 1024 + b] = (float)am;
    p = dout + O_YH + ((size_t)layer * 1024 + b) * 8;
    for (int n = 0; n < 8; ++n) { float oh = (n == am) ? 1.f : 0.f; p[n] = (oh - ys[n]) + ys[n]; }
    dout[O_ZM  + (size_t)layer * 1024 + b] = zm[am];
    dout[O_ZLV + (size_t)layer * 1024 + b] = zlv[am];
    dout[O_ZS  + (size_t)layer * 1024 + b] = z[am];
    nsel[b] = am;
    zsel[b] = z[am];
  }
}

// ================= branch bucketing =================
__global__ void zero_counts_k(int* counts) { if (threadIdx.x < 8) counts[threadIdx.x] = 0; }

__global__ void build_idx_k(const int* nsel, int* counts, int* idx) {
  int b = blockIdx.x * blockDim.x + threadIdx.x;
  if (b >= 1024) return;
  int n = nsel[b];
  int pos = atomicAdd(&counts[n], 1);
  idx[n * 1024 + pos] = b;
}

// relu(out) -> bf16 h
__global__ void h2bf_k(const float* outbuf, unsigned short* hbf) {
  int i = blockIdx.x * blockDim.x + threadIdx.x;
  hbf[i] = f2bf(fmaxf(outbuf[i], 0.f));
}

// ================= stage1: hid = relu(h[sel] @ W1[n] + b1[n]) (bf16 out) =================
// grid (64 mtiles, 8 branches), block 256 = 8 waves; N = 256 -> 16 col tiles,
// 2 per wave, register-blocked (A fragment shared across both WMMAs).
__global__ void stage1_k(const unsigned short* __restrict__ hbf,
                         const unsigned short* __restrict__ w1t,  // (S,NB,256,1024)
                         const float* __restrict__ b1,
                         const int* __restrict__ counts, const int* __restrict__ idx,
                         unsigned short* __restrict__ hidbf, int layer) {
  int n = blockIdx.y;
  int cnt = counts[n];
  int mt = blockIdx.x;
  if (mt * 16 >= cnt) return;
  int lane = threadIdx.x & 31, wv = threadIdx.x >> 5;
  const int* rows = idx + n * 1024 + mt * 16;
  int rloc = lane & 15;
  int grow = (mt * 16 + rloc < cnt) ? rows[rloc] : rows[0];
  const unsigned short* Arow = hbf + (size_t)grow * 1024;
  const unsigned short* Wt = w1t + (size_t)(layer * 8 + n) * 256 * 1024;
  int nt0 = wv * 2;
  v8f acc[2] = {{}, {}};
  for (int k0 = 0; k0 < 1024; k0 += 32) {
    v16bf a = load_frag_row(Arow, k0, lane);
#pragma unroll
    for (int j = 0; j < 2; ++j) {
      v16bf b = load_frag(Wt, 1024, (nt0 + j) * 16, k0, lane);
      acc[j] = wmma_bf16(a, b, acc[j]);
    }
  }
  int mloc = (lane & 16) ? 8 : 0;
#pragma unroll
  for (int j = 0; j < 2; ++j) {
    int col = (nt0 + j) * 16 + (lane & 15);
    float bs = b1[((size_t)layer * 8 + n) * 256 + col];
#pragma unroll
    for (int v = 0; v < 8; ++v) {
      int mi = mt * 16 + mloc + v;
      if (mi < cnt) {
        int bo = rows[mloc + v];
        hidbf[(size_t)bo * 256 + col] = f2bf(fmaxf(acc[j][v] + bs, 0.f));
      }
    }
  }
}

// ================= stage2: out[b] += zsel[b] * (hid[b] @ W2[n] + b2[n]) =================
// grid (64 mtiles, 8 branches), block 256 = 8 waves; N = 1024 -> 64 col tiles,
// 8 per wave, register-blocked (64 accumulator VGPRs, A loaded once per K-step).
__global__ void stage2_k(const unsigned short* __restrict__ hidbf,
                         const unsigned short* __restrict__ w2t,  // (S,NB,1024,256)
                         const float* __restrict__ b2,
                         const int* __restrict__ counts, const int* __restrict__ idx,
                         const float* __restrict__ zsel,
                         float* __restrict__ outbuf, int layer) {
  int n = blockIdx.y;
  int cnt = counts[n];
  int mt = blockIdx.x;
  if (mt * 16 >= cnt) return;
  int lane = threadIdx.x & 31, wv = threadIdx.x >> 5;
  const int* rows = idx + n * 1024 + mt * 16;
  int rloc = lane & 15;
  int grow = (mt * 16 + rloc < cnt) ? rows[rloc] : rows[0];
  const unsigned short* Arow = hidbf + (size_t)grow * 256;
  const unsigned short* Wt = w2t + (size_t)(layer * 8 + n) * 1024 * 256;
  int nt0 = wv * 8;
  v8f acc[8];
#pragma unroll
  for (int j = 0; j < 8; ++j) acc[j] = {};
  for (int k0 = 0; k0 < 256; k0 += 32) {
    v16bf a = load_frag_row(Arow, k0, lane);
#pragma unroll
    for (int j = 0; j < 8; ++j) {
      v16bf b = load_frag(Wt, 256, (nt0 + j) * 16, k0, lane);
      acc[j] = wmma_bf16(a, b, acc[j]);
    }
  }
  int mloc = (lane & 16) ? 8 : 0;
#pragma unroll
  for (int j = 0; j < 8; ++j) {
    int col = (nt0 + j) * 16 + (lane & 15);
    float bs = b2[((size_t)layer * 8 + n) * 1024 + col];
#pragma unroll
    for (int v = 0; v < 8; ++v) {
      int mi = mt * 16 + mloc + v;
      if (mi < cnt) {
        int bo = rows[mloc + v];
        outbuf[(size_t)bo * 1024 + col] += zsel[bo] * (acc[j][v] + bs);
      }
    }
  }
}

// ================= final heads =================
__global__ void head_k(const float* __restrict__ outbuf,
                       const float* __restrict__ fcm_w, const float* __restrict__ fcm_b,
                       const float* __restrict__ fcl_w, const float* __restrict__ fcl_b,
                       float* __restrict__ dout) {
  __shared__ float h[1024];
  int b = blockIdx.x;
  for (int t = threadIdx.x; t < 1024; t += blockDim.x)
    h[t] = fmaxf(outbuf[(size_t)b * 1024 + t], 0.f);
  __syncthreads();
  int j = threadIdx.x;
  if (j < 10) {
    float sm = fcm_b[j], sl = fcl_b[j];
    for (int d = 0; d < 1024; ++d) {
      sm += h[d] * fcm_w[(size_t)d * 10 + j];
      sl += h[d] * fcl_w[(size_t)d * 10 + j];
    }
    dout[O_Z2M  + (size_t)b * 10 + j] = sm;
    dout[O_Z2LV + (size_t)b * 10 + j] = sl;
  }
}

// ================= launch =================
extern "C" void kernel_launch(void* const* d_in, const int* in_sizes, int n_in,
                              void* d_out, int out_size, void* d_ws, size_t ws_size,
                              hipStream_t stream) {
  const float* x       = (const float*)d_in[0];
  const float* gu      = (const float*)d_in[1];
  const float* gn      = (const float*)d_in[2];
  const float* c1w     = (const float*)d_in[3];
  const float* c1b     = (const float*)d_in[4];
  const float* c2w     = (const float*)d_in[5];
  const float* c2b     = (const float*)d_in[6];
  const float* c3w     = (const float*)d_in[7];
  const float* c3b     = (const float*)d_in[8];
  const float* c4w     = (const float*)d_in[9];
  const float* c4b     = (const float*)d_in[10];
  const float* sw_w    = (const float*)d_in[11];
  const float* sw_b    = (const float*)d_in[12];
  const float* sw_w1   = (const float*)d_in[13];
  const float* sw_b1   = (const float*)d_in[14];
  const float* sw_w2   = (const float*)d_in[15];
  const float* sw_b2   = (const float*)d_in[16];
  const float* fcm_w   = (const float*)d_in[17];
  const float* fcm_b   = (const float*)d_in[18];
  const float* fcl_w   = (const float*)d_in[19];
  const float* fcl_b   = (const float*)d_in[20];
  float* dout = (float*)d_out;

  char* w = (char*)d_ws;
  size_t o = 0;
  auto take = [&](size_t bytes) -> char* {
    char* p = w + o;
    o += (bytes + 255) & ~(size_t)255;
    return p;
  };
  unsigned short* patch  = (unsigned short*)take((size_t)262144 * 512 * 2); // max conv2 patches
  float* act1            = (float*)take((size_t)1024 * 32 * 32 * 32 * 4);
  float* act2            = (float*)take((size_t)1024 * 32 * 16 * 16 * 4);
  float* act3            = (float*)take((size_t)1024 * 64 * 8 * 8 * 4);
  float* outbuf          = (float*)take((size_t)1024 * 1024 * 4);
  unsigned short* wc1    = (unsigned short*)take((size_t)32 * 64 * 2);
  unsigned short* wc2    = (unsigned short*)take((size_t)32 * 512 * 2);
  unsigned short* wc3    = (unsigned short*)take((size_t)64 * 512 * 2);
  unsigned short* wc4    = (unsigned short*)take((size_t)64 * 1024 * 2);
  unsigned short* w1t    = (unsigned short*)take((size_t)4 * 8 * 256 * 1024 * 2);
  unsigned short* w2t    = (unsigned short*)take((size_t)4 * 8 * 1024 * 256 * 2);
  unsigned short* hbf    = (unsigned short*)take((size_t)1024 * 1024 * 2);
  unsigned short* hidbf  = (unsigned short*)take((size_t)1024 * 256 * 2);
  int*   nsel            = (int*)take(1024 * 4);
  float* zsel            = (float*)take(1024 * 4);
  int*   counts          = (int*)take(32);
  int*   idxl            = (int*)take(8 * 1024 * 4);

  // ---- weight prep ----
  wprep_conv_k<<<(32 * 64 + 255) / 256, 256, 0, stream>>>(c1w, wc1, 32, 48, 64);
  wprep_conv_k<<<(32 * 512 + 255) / 256, 256, 0, stream>>>(c2w, wc2, 32, 512, 512);
  wprep_conv_k<<<(64 * 512 + 255) / 256, 256, 0, stream>>>(c3w, wc3, 64, 512, 512);
  wprep_conv_k<<<(64 * 1024 + 255) / 256, 256, 0, stream>>>(c4w, wc4, 64, 1024, 1024);
  w1prep_k<<<(4 * 8 * 256 * 1024) / 256, 256, 0, stream>>>(sw_w1, w1t);
  w2prep_k<<<(4 * 8 * 1024 * 256) / 256, 256, 0, stream>>>(sw_w2, w2t);

  // ---- conv pipeline (im2col + bf16 WMMA GEMM, NCHW epilogue) ----
  im2col_k<<<(int)(((size_t)1024 * 32 * 32 * 64) / 256), 256, 0, stream>>>(x, patch, 3, 64, 64, 32, 32, 64);
  conv_gemm_k<2><<<8192, 256, 0, stream>>>(patch, wc1, c1b, act1, 64, 1024, 1);

  im2col_k<<<(int)(((size_t)1024 * 16 * 16 * 512) / 256), 256, 0, stream>>>(act1, patch, 32, 32, 32, 16, 16, 512);
  conv_gemm_k<2><<<2048, 256, 0, stream>>>(patch, wc2, c2b, act2, 512, 256, 1);

  im2col_k<<<(int)(((size_t)1024 * 8 * 8 * 512) / 256), 256, 0, stream>>>(act2, patch, 32, 16, 16, 8, 8, 512);
  conv_gemm_k<4><<<512, 256, 0, stream>>>(patch, wc3, c3b, act3, 512, 64, 1);

  im2col_k<<<(int)(((size_t)1024 * 4 * 4 * 1024) / 256), 256, 0, stream>>>(act3, patch, 64, 8, 8, 4, 4, 1024);
  conv_gemm_k<4><<<128, 256, 0, stream>>>(patch, wc4, c4b, outbuf, 1024, 16, 0); // no relu: raw "out"

  // ---- stochastic switch layers (selected-branch gather GEMMs) ----
  for (int i = 0; i < 4; ++i) {
    h2bf_k<<<4096, 256, 0, stream>>>(outbuf, hbf);
    ctrl_k<<<1024, 128, 0, stream>>>(outbuf, sw_w, sw_b, gu, gn, dout, i, nsel, zsel);
    zero_counts_k<<<1, 32, 0, stream>>>(counts);
    build_idx_k<<<4, 256, 0, stream>>>(nsel, counts, idxl);
    stage1_k<<<dim3(64, 8), 256, 0, stream>>>(hbf, w1t, sw_b1, counts, idxl, hidbf, i);
    stage2_k<<<dim3(64, 8), 256, 0, stream>>>(hidbf, w2t, sw_b2, counts, idxl, zsel, outbuf, i);
  }

  // ---- heads ----
  head_k<<<1024, 128, 0, stream>>>(outbuf, fcm_w, fcm_b, fcl_w, fcl_b, dout);
}
